// GraphAttentionNet_61907658605026
// MI455X (gfx1250) — matrix operand order repaired and checked by reading, not compiled, not run on hardware
//
#include <hip/hip_runtime.h>
#include <hip/hip_bf16.h>

// ---------------------------------------------------------------------------
// 3-layer GATConv for MI455X (gfx1250, wave32).
//   GEMM (node projection)     : WMMA f32<-f16 16x16x32, LDS-tiled,
//                                vectorized branch-free staging.
//   Edge softmax + aggregation : coalesced float4 gather + f32 global atomics
//                                (memory-bound part, ~2.6 GB @ 23.3 TB/s).
// ---------------------------------------------------------------------------

typedef __attribute__((ext_vector_type(16))) _Float16 v16h;
typedef __attribute__((ext_vector_type(8)))  float    v8f;

#define NNODES 20000
#define NEDGES 256000
#define NEG_SLOPE 0.2f

#define BMT 128        // block tile M (8 waves x 16 rows)
#define BN  32         // block tile N (each wave does all 32 cols -> 2 WMMAs)
#define BK  32         // K step
#define LDP 36         // padded row stride in halves (72 B: 8B-aligned, conflict-free)

// ---------------------------------------------------------------------------
// Tiled GEMM:  C[M,N] = A[M,K] @ B[K,N]   (f32 in, f16 WMMA, f32 accumulate)
// 256 threads = 8 waves; wave w owns rows [w*16, w*16+16) x all 32 cols.
// ---------------------------------------------------------------------------
__global__ __launch_bounds__(256) void gemm_wmma_kernel(
    const float* __restrict__ A, const float* __restrict__ B,
    float* __restrict__ C, int M, int K, int N)
{
    __shared__ _Float16 As[BMT][LDP];
    __shared__ _Float16 Bs[BN][LDP];     // column-major: Bs[n][k]

    const int tid  = threadIdx.x;
    const int lane = tid & 31;
    const int wave = tid >> 5;       // 0..7 : M sub-tile
    const int hi   = lane >> 4;      // lane half (0/1)
    const int l15  = lane & 15;
    const int m0   = blockIdx.x * BMT;
    const int n0   = blockIdx.y * BN;

    const bool mOK = (m0 + BMT) <= M;
    const bool nOK = (n0 + BN)  <= N;

    v8f acc0 = {0.f, 0.f, 0.f, 0.f, 0.f, 0.f, 0.f, 0.f};
    v8f acc1 = {0.f, 0.f, 0.f, 0.f, 0.f, 0.f, 0.f, 0.f};

    for (int k0 = 0; k0 < K; k0 += BK) {
        const bool kOK = (k0 + BK) <= K;

        // ---- stage A tile (BMT x BK), f32 -> f16 ------------------------
        if (mOK && kOK) {
            // fast path: float4 loads (coalesced), 8B LDS stores
            #pragma unroll
            for (int it = 0; it < 4; ++it) {
                const int r = (tid >> 3) + it * 32;
                const int c = (tid & 7) * 4;
                const float4 v = *(const float4*)(A + (size_t)(m0 + r) * K + (k0 + c));
                _Float16* p = &As[r][c];
                p[0] = (_Float16)v.x; p[1] = (_Float16)v.y;
                p[2] = (_Float16)v.z; p[3] = (_Float16)v.w;
            }
        } else {
            // boundary path, phase 1: clamped branchless loads into regs
            float va[16];
            #pragma unroll
            for (int it = 0; it < 16; ++it) {
                const int i = tid + it * 256;
                const int r = i >> 5, c = i & 31;
                const int gm = m0 + r, gk = k0 + c;
                const bool ok = (gm < M) && (gk < K);
                const size_t idx = ok ? ((size_t)gm * K + gk) : 0;
                const float v = A[idx];
                va[it] = ok ? v : 0.0f;
            }
            // phase 2: convert + store (loads batched under one wait)
            #pragma unroll
            for (int it = 0; it < 16; ++it) {
                const int i = tid + it * 256;
                As[i >> 5][i & 31] = (_Float16)va[it];
            }
        }

        // ---- stage B tile (BK x BN) into column-major Bs[n][k] ----------
        if (nOK && kOK) {
            // wave w loads rows k = 4w..4w+3; each row is a coalesced 128B load
            const int kk = (tid >> 5) * 4;
            const int nn = tid & 31;
            const float* bp = B + (size_t)(k0 + kk) * N + (n0 + nn);
            const float b0 = bp[0];
            const float b1 = bp[(size_t)N];
            const float b2 = bp[(size_t)2 * N];
            const float b3 = bp[(size_t)3 * N];
            _Float16* p = &Bs[nn][kk];
            p[0] = (_Float16)b0; p[1] = (_Float16)b1;
            p[2] = (_Float16)b2; p[3] = (_Float16)b3;
        } else {
            float vb[4];
            #pragma unroll
            for (int it = 0; it < 4; ++it) {
                const int i = tid + it * 256;
                const int r = i >> 5, c = i & 31;   // r = k, c = n
                const int gk = k0 + r, gn = n0 + c;
                const bool ok = (gk < K) && (gn < N);
                const size_t idx = ok ? ((size_t)gk * N + gn) : 0;
                const float v = B[idx];
                vb[it] = ok ? v : 0.0f;
            }
            #pragma unroll
            for (int it = 0; it < 4; ++it) {
                const int i = tid + it * 256;
                Bs[i & 31][i >> 5] = (_Float16)vb[it];
            }
        }
        __syncthreads();

        // ---- fragments: contiguous K order k = 16*hi + j, same in A and B --
        v16h af, bf0, bf1;
        const int am = wave * 16 + l15;               // A row (M)
        #pragma unroll
        for (int j = 0; j < 8; ++j) {                 // lanes 0-15: K 0-15
            af[j]     = As[am][hi * 16 + j];          // lanes16-31: K 16-31
            af[j + 8] = As[am][hi * 16 + 8 + j];
        }
        #pragma unroll
        for (int j = 0; j < 16; ++j) {                // lanes 0-15: K 0-15
            bf0[j] = Bs[l15][hi * 16 + j];            // lanes16-31: K 16-31
            bf1[j] = Bs[16 + l15][hi * 16 + j];
        }

        acc0 = __builtin_amdgcn_wmma_f32_16x16x32_f16(
            false, af, false, bf0, (short)0, acc0, false, false);
        acc1 = __builtin_amdgcn_wmma_f32_16x16x32_f16(
            false, af, false, bf1, (short)0, acc1, false, false);
        __syncthreads();
    }

    // ---- store per 32-bit C/D layout: VGPR r -> M = r (+8 upper half) ---
    const int gmBase = m0 + wave * 16 + hi * 8;
    const int gnA = n0 + l15;
    const int gnB = n0 + 16 + l15;
    #pragma unroll
    for (int r = 0; r < 8; ++r) {
        const int gm = gmBase + r;
        if (gm < M) {
            if (gnA < N) C[(size_t)gm * N + gnA] = acc0[r];
            if (gnB < N) C[(size_t)gm * N + gnB] = acc1[r];
        }
    }
}

// ---------------------------------------------------------------------------
// Per-(node,head) attention logits: el = <h, al>, er = <h, ar>
// ---------------------------------------------------------------------------
__global__ void attn_scores_kernel(const float* __restrict__ h,
                                   const float* __restrict__ al,
                                   const float* __restrict__ ar,
                                   float* __restrict__ el, float* __restrict__ er,
                                   int nNodes, int H, int F)
{
    int t = blockIdx.x * blockDim.x + threadIdx.x;
    if (t >= nNodes * H) return;
    int node = t / H, head = t % H;
    const float* hp  = h  + (size_t)node * H * F + (size_t)head * F;
    const float* alp = al + (size_t)head * F;
    const float* arp = ar + (size_t)head * F;
    float sl = 0.f, sr = 0.f;
    for (int f = 0; f < F; ++f) {
        float v = hp[f];
        sl += v * alp[f];
        sr += v * arp[f];
    }
    el[t] = sl;
    er[t] = sr;
}

// Classic float atomic-max via int punning (valid with 0xFF... init pattern).
__device__ inline void atomicMaxF(float* addr, float val)
{
    if (val >= 0.f) atomicMax((int*)addr, __float_as_int(val));
    else            atomicMin((unsigned int*)addr, __float_as_uint(val));
}

// ---------------------------------------------------------------------------
// Pass 1: edge score = leaky_relu(el[src] + er[dst]); segment max over dst.
// ---------------------------------------------------------------------------
__global__ void edge_score_max_kernel(const int* __restrict__ src,
                                      const int* __restrict__ dst,
                                      const float* __restrict__ el,
                                      const float* __restrict__ er,
                                      float* __restrict__ esc,
                                      float* __restrict__ emax,
                                      int E, int H)
{
    int t = blockIdx.x * blockDim.x + threadIdx.x;
    if (t >= E * H) return;
    int e = t / H, hd = t % H;
    int s = src[e], d = dst[e];
    float v = el[s * H + hd] + er[d * H + hd];
    v = v > 0.f ? v : v * NEG_SLOPE;
    esc[t] = v;
    atomicMaxF(&emax[d * H + hd], v);
}

// ---------------------------------------------------------------------------
// Pass 2: ee = exp(e - emax[dst]); segment sum over dst. (esc updated in place)
// ---------------------------------------------------------------------------
__global__ void edge_exp_sum_kernel(const int* __restrict__ dst,
                                    float* __restrict__ esc,
                                    const float* __restrict__ emax,
                                    float* __restrict__ den,
                                    int E, int H)
{
    int t = blockIdx.x * blockDim.x + threadIdx.x;
    if (t >= E * H) return;
    int e = t / H, hd = t % H;
    int d = dst[e];
    float ee = __expf(esc[t] - emax[d * H + hd]);
    esc[t] = ee;
    atomicAdd(&den[d * H + hd], ee);
}

// ---------------------------------------------------------------------------
// Pass 3: aggr[dst] += h[src] * alpha   (float4 coalesced; f32 atomics)
// ---------------------------------------------------------------------------
__global__ void edge_aggregate_kernel(const int* __restrict__ src,
                                      const int* __restrict__ dst,
                                      const float* __restrict__ h,
                                      const float* __restrict__ esc,
                                      const float* __restrict__ den,
                                      float* __restrict__ aggr,
                                      int E, int H, int F)
{
    const int HF  = H * F;
    const int HF4 = HF >> 2;
    int t = blockIdx.x * blockDim.x + threadIdx.x;
    if (t >= E * HF4) return;
    int e    = t / HF4;
    int col  = (t % HF4) << 2;     // F % 4 == 0 for all layers -> single head
    int head = col / F;
    int s = src[e], d = dst[e];
    float alpha = esc[e * H + head] / (den[d * H + head] + 1e-9f);
    const float4 hv = *(const float4*)(h + (size_t)s * HF + col);
    float* out = aggr + (size_t)d * HF + col;
    atomicAdd(out + 0, hv.x * alpha);
    atomicAdd(out + 1, hv.y * alpha);
    atomicAdd(out + 2, hv.z * alpha);
    atomicAdd(out + 3, hv.w * alpha);
}

// ---------------------------------------------------------------------------
// Epilogues (in place): x = elu(x + b)  /  x = sigmoid(x + b)
// ---------------------------------------------------------------------------
__global__ void bias_elu_kernel(float* __restrict__ x, const float* __restrict__ b,
                                int n, int cols)
{
    int t = blockIdx.x * blockDim.x + threadIdx.x;
    if (t >= n) return;
    float v = x[t] + b[t % cols];
    x[t] = v > 0.f ? v : (__expf(v) - 1.0f);
}

__global__ void bias_sigmoid_kernel(float* __restrict__ x, const float* __restrict__ b,
                                    int n, int cols)
{
    int t = blockIdx.x * blockDim.x + threadIdx.x;
    if (t >= n) return;
    float v = x[t] + b[t % cols];
    x[t] = 1.0f / (1.0f + __expf(-v));
}

// ---------------------------------------------------------------------------
// Host-side layer sequencer (all launches on `stream`, capture-safe).
// ---------------------------------------------------------------------------
static void gat_layer(const float* in, const float* W, const float* al,
                      const float* ar, const float* b, int K, int H, int F,
                      float* hproj, float* el, float* er, float* esc,
                      float* emax, float* den, float* aggr,
                      const int* src, const int* dst, int act,
                      hipStream_t stream)
{
    const int Nn = NNODES, E = NEDGES;
    const int HF = H * F;

    dim3 gg((Nn + BMT - 1) / BMT, (HF + BN - 1) / BN);
    gemm_wmma_kernel<<<gg, 256, 0, stream>>>(in, W, hproj, Nn, K, HF);

    int nh = Nn * H;
    attn_scores_kernel<<<(nh + 255) / 256, 256, 0, stream>>>(hproj, al, ar, el, er, Nn, H, F);

    hipMemsetAsync(emax, 0xFF, (size_t)nh * 4, stream);   // 0xFF... works for atomicMaxF
    hipMemsetAsync(den,  0x00, (size_t)nh * 4, stream);

    int eh = E * H;
    edge_score_max_kernel<<<(eh + 255) / 256, 256, 0, stream>>>(src, dst, el, er, esc, emax, E, H);
    edge_exp_sum_kernel<<<(eh + 255) / 256, 256, 0, stream>>>(dst, esc, emax, den, E, H);

    hipMemsetAsync(aggr, 0x00, (size_t)Nn * HF * 4, stream);
    int tot4 = E * (HF / 4);
    edge_aggregate_kernel<<<(tot4 + 255) / 256, 256, 0, stream>>>(src, dst, hproj, esc, den, aggr, E, H, F);

    int nHF = Nn * HF;
    if (act == 0)
        bias_elu_kernel<<<(nHF + 255) / 256, 256, 0, stream>>>(aggr, b, nHF, HF);
    else
        bias_sigmoid_kernel<<<(nHF + 255) / 256, 256, 0, stream>>>(aggr, b, nHF, HF);
}

// Workspace layout (bytes; all offsets 256-aligned). Total ~165.4 MB.
#define OFF_H    ((size_t)0)            // 64,000,000 : projected features (max 20000*800*4)
#define OFF_B1   ((size_t)64000000)     // 32,000,000 : aggr1 -> act1 (layer-2 input)
#define OFF_B2   ((size_t)96000000)     // 64,000,000 : aggr2 -> act2 (layer-3 input)
#define OFF_ESC  ((size_t)160000000)    //  4,096,000 : per-edge scores (E*Hmax*4)
#define OFF_EMAX ((size_t)164096000)    //    320,000
#define OFF_DEN  ((size_t)164416000)    //    320,000
#define OFF_EL   ((size_t)164736000)    //    320,000
#define OFF_ER   ((size_t)165056000)    //    320,000

extern "C" void kernel_launch(void* const* d_in, const int* in_sizes, int n_in,
                              void* d_out, int out_size, void* d_ws, size_t ws_size,
                              hipStream_t stream)
{
    (void)in_sizes; (void)n_in; (void)out_size; (void)ws_size;

    const float* x   = (const float*)d_in[0];
    const int*   src = (const int*)  d_in[1];
    const int*   dst = (const int*)  d_in[2];
    const float* W1  = (const float*)d_in[3];
    const float* al1 = (const float*)d_in[4];
    const float* ar1 = (const float*)d_in[5];
    const float* b1  = (const float*)d_in[6];
    const float* W2  = (const float*)d_in[7];
    const float* al2 = (const float*)d_in[8];
    const float* ar2 = (const float*)d_in[9];
    const float* b2  = (const float*)d_in[10];
    const float* W3  = (const float*)d_in[11];
    const float* al3 = (const float*)d_in[12];
    const float* ar3 = (const float*)d_in[13];
    const float* b3  = (const float*)d_in[14];

    char* ws = (char*)d_ws;
    float* hproj = (float*)(ws + OFF_H);
    float* buf1  = (float*)(ws + OFF_B1);
    float* buf2  = (float*)(ws + OFF_B2);
    float* esc   = (float*)(ws + OFF_ESC);
    float* emax  = (float*)(ws + OFF_EMAX);
    float* den   = (float*)(ws + OFF_DEN);
    float* el    = (float*)(ws + OFF_EL);
    float* er    = (float*)(ws + OFF_ER);
    float* out   = (float*)d_out;

    // Layer 1: 128 -> 4 x 100, ELU
    gat_layer(x,    W1, al1, ar1, b1, 128, 4, 100, hproj, el, er, esc, emax, den, buf1, src, dst, 0, stream);
    // Layer 2: 400 -> 4 x 200, ELU
    gat_layer(buf1, W2, al2, ar2, b2, 400, 4, 200, hproj, el, er, esc, emax, den, buf2, src, dst, 0, stream);
    // Layer 3: 800 -> 1 x 64, sigmoid (aggregate straight into d_out)
    gat_layer(buf2, W3, al3, ar3, b3, 800, 1, 64,  hproj, el, er, esc, emax, den, out,  src, dst, 1, stream);
}
